// GRUSimpleModel_1219770712112
// MI455X (gfx1250) — matrix operand order repaired and checked by reading, not compiled, not run on hardware
//
#include <hip/hip_runtime.h>

typedef __attribute__((ext_vector_type(16))) _Float16 v16h;
typedef __attribute__((ext_vector_type(8)))  _Float16 v8h;
typedef __attribute__((ext_vector_type(8)))  float    v8f;

#define NB 256      // batch
#define NT 2048     // time
#define NF 37       // features
#define TF 111      // 3*F
#define KP 128      // padded K for projection GEMM
#define NU 32       // units
#define TU 96       // 3*U
#define ND 16       // demo dim

// Permuted XZ layout: element (b, t, col) lives at
//   (((t*16 + (b>>4))*2 + ((col>>4)&1))*32 + (col&15) + (((b>>3)&1)<<4))*24
//     + (col>>5)*8 + (b&7)
// i.e. per (t, batch-tile, col-half wave, lane): 24 f16 = 48 contiguous bytes,
// ordered [gate g][row r].  g = col>>5 (z,r,h), q = (col>>4)&1 (col half).
#define XZ_STEP (16 * 2 * 32 * 24)   // elements per time step (all tiles)

__device__ __forceinline__ float sigmoidf_(float x) {
    return 1.f / (1.f + __expf(-x));
}

// ---------------------------------------------------------------------------
// Kernel 1: h0 = relu(demo @ W1 + b1) @ W2 + b2    (tiny: 256x16x32, 256x32x32)
// ---------------------------------------------------------------------------
__global__ __launch_bounds__(256) void k_h0(
    const float* __restrict__ demo, const float* __restrict__ W1,
    const float* __restrict__ b1, const float* __restrict__ W2,
    const float* __restrict__ b2, float* __restrict__ h0) {
  int i = blockIdx.x * blockDim.x + threadIdx.x;
  if (i >= NB) return;
  float d[ND];
#pragma unroll
  for (int k = 0; k < ND; ++k) d[k] = demo[i * ND + k];
  float hid[NU];
#pragma unroll
  for (int u = 0; u < NU; ++u) {
    float s = b1[u];
#pragma unroll
    for (int k = 0; k < ND; ++k) s += d[k] * W1[k * NU + u];
    hid[u] = s > 0.f ? s : 0.f;
  }
#pragma unroll
  for (int u = 0; u < NU; ++u) {
    float s = b2[u];
#pragma unroll
    for (int k = 0; k < NU; ++k) s += hid[k] * W2[k * NU + u];
    h0[i * NU + u] = s;
  }
}

// ---------------------------------------------------------------------------
// Kernel 2: XZ = concat(values,meas,dt) @ kernel + b_in, written in the
// scan-fragment-permuted f16 layout. GEMM M=B*T=524288, K=111(pad128), N=96.
// One 16-row M-tile per wave, 8 waves/block, v_wmma_f32_16x16x32_f16.
// ---------------------------------------------------------------------------
__global__ __launch_bounds__(256) void k_proj(
    const float* __restrict__ values, const int* __restrict__ meas,
    const float* __restrict__ dt, const float* __restrict__ kern,
    const float* __restrict__ bias, _Float16* __restrict__ XZp) {
  __shared__ _Float16 ldsK[TU * KP];      // kernel^T, padded: [col][k]  (24 KB)
  __shared__ _Float16 ldsX[8 * 16 * KP];  // per-wave 16x128 x-tile      (32 KB)

  const int tid = threadIdx.x, lane = tid & 31, w = tid >> 5;

  // Stage kernel transposed to f16, zero-pad K to 128.
  for (int idx = tid; idx < TU * KP; idx += 256) {
    int c = idx >> 7, k = idx & (KP - 1);
    ldsK[idx] = (k < TF) ? (_Float16)kern[k * TU + c] : (_Float16)0.f;
  }
  __syncthreads();

  const int mt = blockIdx.x * 8 + w;   // M-tile index
  const int m0 = mt * 16;
  const int b  = m0 >> 11;             // / NT  (tile never crosses a b row)
  const int t0 = m0 & (NT - 1);
  _Float16* xw = &ldsX[w * 16 * KP];

  // Stage 16 rows of concat(values, meas, dt) into LDS as f16.
  for (int idx = lane; idx < 16 * NF; idx += 32) {
    int r = idx / NF, k = idx - r * NF;
    size_t g = (size_t)(b * NT + t0 + r) * NF + k;
    xw[r * KP + k]          = (_Float16)values[g];
    xw[r * KP + NF + k]     = (_Float16)(float)meas[g];
    xw[r * KP + 2 * NF + k] = (_Float16)dt[g];
  }
  for (int idx = lane; idx < 16 * (KP - TF); idx += 32) {
    int r = idx / (KP - TF), k = TF + (idx - r * (KP - TF));
    xw[r * KP + k] = (_Float16)0.f;
  }

  const int rA = lane & 15, ka = (lane >> 4) * 16;  // A-frag row / K-base
  const int cB = lane & 15, kb = (lane >> 4) * 16;  // B-frag col / K-base

  v8f acc[6];
#pragma unroll
  for (int n = 0; n < 6; ++n) {
    float bv = bias[n * 16 + cB];  // input bias (bias[0])
#pragma unroll
    for (int r = 0; r < 8; ++r) acc[n][r] = bv;
  }

#pragma unroll
  for (int kk = 0; kk < 4; ++kk) {
    v8h a0 = *(const v8h*)&xw[rA * KP + kk * 32 + ka];
    v8h a1 = *(const v8h*)&xw[rA * KP + kk * 32 + ka + 8];
    v16h a;
#pragma unroll
    for (int i = 0; i < 8; ++i) { a[i] = a0[i]; a[8 + i] = a1[i]; }
#pragma unroll
    for (int n = 0; n < 6; ++n) {
      v8h q0 = *(const v8h*)&ldsK[(n * 16 + cB) * KP + kk * 32 + kb];
      v8h q1 = *(const v8h*)&ldsK[(n * 16 + cB) * KP + kk * 32 + kb + 8];
      v16h bf;
#pragma unroll
      for (int i = 0; i < 8; ++i) { bf[i] = q0[i]; bf[8 + i] = q1[i]; }
      acc[n] = __builtin_amdgcn_wmma_f32_16x16x32_f16(
          false, a, false, bf, (short)0, acc[n], false, false);
    }
  }

  // Scatter-store into scan-permuted layout. b is wave-invariant, so the
  // scan-side tile / lane / row-subindex are loop-invariant; only t varies.
  const int rhi    = (lane >> 4) * 8;
  const int tile   = b >> 4;
  const int bi     = b & 15;
  const int lane_s = cB + ((bi >> 3) << 4);
  const int r_s    = bi & 7;
#pragma unroll
  for (int r = 0; r < 8; ++r) {
    int t = t0 + r + rhi;
    // base for q=0; q=1 is +32*24 elements
    size_t base0 = (((size_t)(t * 16 + tile) * 2) * 32 + lane_s) * 24 + r_s;
#pragma unroll
    for (int n = 0; n < 6; ++n) {
      size_t addr = base0 + (size_t)(n & 1) * (32 * 24) + (n >> 1) * 8;
      XZp[addr] = (_Float16)acc[n][r];
    }
  }
}

// ---------------------------------------------------------------------------
// Kernel 3: sequential GRU scan. One 16-batch-row tile per block; 2 waves per
// block split the step by column half q: wave q owns gate tiles {q,2+q,4+q}
// and state columns h[:, q*16..q*16+15].  Per step/wave: 3x b128 xz loads,
// 3 WMMAs, 24 transcendentals; full h shared via ping-pong LDS buffer with
// one __syncthreads per step.  xz+b_rec folded into WMMA C for z/r gates.
// Prefix mask -> early exit at tile max length.
// ---------------------------------------------------------------------------
__global__ __launch_bounds__(64) void k_scan(
    const _Float16* __restrict__ XZp, const float* __restrict__ h0,
    const float* __restrict__ reck, const float* __restrict__ bias,
    const int* __restrict__ lengths, const float* __restrict__ Wo,
    const float* __restrict__ bo, float* __restrict__ out) {
  __shared__ _Float16 ldsH[2 * 16 * NU];  // ping-pong h buffer (2 KB)
  __shared__ float    ldsP[2 * 16];       // head partial sums

  const int lane = threadIdx.x & 31;
  const int q    = threadIdx.x >> 5;      // column-half / wave id
  const int c  = lane & 15;               // column within 16-wide N tile
  const int hi = lane >> 4;               // row-half select (C layout)
  const int rA = lane & 15, ka = hi * 16; // A-frag addressing
  const int kb = hi * 16;                 // B-frag K base
  const int b0 = blockIdx.x * 16;

  // Per-(wave,lane) contiguous xz stream: 48 B per step, order [g][r].
  const _Float16* xzbase =
      &XZp[(((size_t)blockIdx.x * 2 + q) * 32 + lane) * 24];

  // rec_kernel B-fragments for this wave's gate tiles n = g*2+q.
  v16h rk[3];
#pragma unroll
  for (int g = 0; g < 3; ++g)
#pragma unroll
    for (int j = 0; j < 16; ++j)
      rk[g][j] = (_Float16)reck[(kb + j) * TU + (g * 2 + q) * 16 + c];

  float brec[3];
#pragma unroll
  for (int g = 0; g < 3; ++g) brec[g] = bias[TU + (g * 2 + q) * 16 + c];

  // Lengths of this lane's 8 rows; tile max (same in both waves).
  int lenv[8];
  int lmax = 0;
#pragma unroll
  for (int r = 0; r < 8; ++r) {
    lenv[r] = lengths[b0 + r + hi * 8];
    lmax = max(lmax, lenv[r]);
  }
#pragma unroll
  for (int m = 16; m >= 1; m >>= 1) lmax = max(lmax, __shfl_xor(lmax, m, 32));

  // Initial state: this wave's columns, C layout: h[r] = h0[row][q*16+c].
  float h[8];
#pragma unroll
  for (int r = 0; r < 8; ++r) {
    int row = b0 + r + hi * 8;
    h[r] = h0[row * NU + q * 16 + c];
    ldsH[(r + hi * 8) * NU + q * 16 + c] = (_Float16)h[r];  // buffer 0
  }
  __syncthreads();

  for (int t = 0; t < lmax; ++t) {
    const _Float16* xzp = xzbase + (size_t)t * XZ_STEP;

    // Prefetch next step's 48 B (fixed +49152 B stride; L2-resident target).
    if (t + 1 < NT)
      __builtin_prefetch((const char*)xzp + XZ_STEP * sizeof(_Float16), 0, 1);

    // 24 contiguous f16: xzv[g][r], 3 x b128 loads.
    v8h xzv[3];
    const v8h* xv = (const v8h*)xzp;
#pragma unroll
    for (int g = 0; g < 3; ++g) xzv[g] = xv[g];

    // A fragment (full 16x32 h) from current LDS buffer.
    const _Float16* hb = &ldsH[(t & 1) * 16 * NU];
    v8h a0 = *(const v8h*)&hb[rA * NU + ka];
    v8h a1 = *(const v8h*)&hb[rA * NU + ka + 8];
    v16h a;
#pragma unroll
    for (int i = 0; i < 8; ++i) { a[i] = a0[i]; a[8 + i] = a1[i]; }

    // rec for gate tiles; z/r fold xz + b_rec into the accumulator init.
    v8f accZ, accR, accH;
#pragma unroll
    for (int r = 0; r < 8; ++r) {
      accZ[r] = (float)xzv[0][r] + brec[0];
      accR[r] = (float)xzv[1][r] + brec[1];
      accH[r] = brec[2];
    }
    accZ = __builtin_amdgcn_wmma_f32_16x16x32_f16(
        false, a, false, rk[0], (short)0, accZ, false, false);
    accR = __builtin_amdgcn_wmma_f32_16x16x32_f16(
        false, a, false, rk[1], (short)0, accR, false, false);
    accH = __builtin_amdgcn_wmma_f32_16x16x32_f16(
        false, a, false, rk[2], (short)0, accH, false, false);

    // Gates + masked state update (prefix mask: t < length).
    _Float16* hbn = &ldsH[((t + 1) & 1) * 16 * NU];
#pragma unroll
    for (int r = 0; r < 8; ++r) {
      float z  = sigmoidf_(accZ[r]);
      float rr = sigmoidf_(accR[r]);
      float hh = tanhf((float)xzv[2][r] + rr * accH[r]);
      float hn = z * h[r] + (1.f - z) * hh;
      h[r] = (t < lenv[r]) ? hn : h[r];
      hbn[(r + hi * 8) * NU + q * 16 + c] = (_Float16)h[r];
    }
    __syncthreads();
  }

  // Head: out[row] = sigmoid(h_last . Wo + bo); cross-wave via LDS partials.
  float woQ = Wo[q * 16 + c], bov = bo[0];
#pragma unroll
  for (int r = 0; r < 8; ++r) {
    float p = h[r] * woQ;
#pragma unroll
    for (int m = 8; m >= 1; m >>= 1) p += __shfl_xor(p, m, 32);
    if (c == 0) ldsP[q * 16 + hi * 8 + r] = p;
  }
  __syncthreads();
  if (threadIdx.x < 16) {
    int row = threadIdx.x;
    out[b0 + row] = sigmoidf_(ldsP[row] + ldsP[16 + row] + bov);
  }
}

// ---------------------------------------------------------------------------
extern "C" void kernel_launch(void* const* d_in, const int* in_sizes, int n_in,
                              void* d_out, int out_size, void* d_ws, size_t ws_size,
                              hipStream_t stream) {
  const float* demo    = (const float*)d_in[0];
  const float* dt      = (const float*)d_in[1];
  const float* values  = (const float*)d_in[2];
  const int*   meas    = (const int*)d_in[3];
  const int*   lengths = (const int*)d_in[4];
  const float* W1      = (const float*)d_in[5];
  const float* b1      = (const float*)d_in[6];
  const float* W2      = (const float*)d_in[7];
  const float* b2      = (const float*)d_in[8];
  const float* kern    = (const float*)d_in[9];
  const float* reck    = (const float*)d_in[10];
  const float* bias    = (const float*)d_in[11];
  const float* Wo      = (const float*)d_in[12];
  const float* bo      = (const float*)d_in[13];
  float* out = (float*)d_out;

  // Workspace layout: h0 [256x32 f32] then permuted XZ [B*T*96 f16] (~96 MB).
  float*    h0v = (float*)d_ws;
  _Float16* XZp = (_Float16*)((char*)d_ws + (size_t)NB * NU * sizeof(float));

  k_h0<<<1, 256, 0, stream>>>(demo, W1, b1, W2, b2, h0v);
  k_proj<<<(NB * NT / 16) / 8, 256, 0, stream>>>(values, meas, dt, kern, bias, XZp);
  k_scan<<<NB / 16, 64, 0, stream>>>(XZp, h0v, reck, bias, lengths, Wo, bo, out);
}
